// GOFADecoderLayer_55963423866999
// MI455X (gfx1250) — compile-verified, matrix operand order checked by reading, not run on hardware
//
#include <hip/hip_runtime.h>

// ---------------------------------------------------------------------------
// GOFA decoder layer for MI455X (gfx1250, wave32, WMMA).
// All GEMMs run on v_wmma_f32_16x16x32_bf16 with the shared A tile staged
// into LDS via the CDNA5 async-copy path (global_load_async_to_lds_b64,
// ASYNCcnt, double-buffered).  Softmax/segment ops in f32 with L2 atomics.
// ---------------------------------------------------------------------------

typedef __bf16 bf16;
typedef bf16          v16bf  __attribute__((ext_vector_type(16)));
typedef float         v8f    __attribute__((ext_vector_type(8)));
typedef unsigned int  u32x4  __attribute__((ext_vector_type(4)));

static constexpr int N_    = 1024;
static constexpr int E_    = 8192;
static constexpr int QL_   = 8;
static constexpr int HID_  = 1024;
static constexpr int NH_   = 16;
static constexpr int NKV_  = 8;
static constexpr int HD_   = 64;
static constexpr int INTER_= 4096;
static constexpr int KP_   = 2 * QL_;   // 16 key positions per edge

// ---------------------------------------------------------------------------
// Order-preserving f32 <-> u32 encoding for atomicMax-based segment max.
// ---------------------------------------------------------------------------
__device__ __forceinline__ unsigned enc_ord(float f) {
  unsigned i = __float_as_uint(f);
  return (i & 0x80000000u) ? ~i : (i | 0x80000000u);
}
__device__ __forceinline__ float dec_ord(unsigned k) {
  unsigned i = (k & 0x80000000u) ? (k ^ 0x80000000u) : ~k;
  return __uint_as_float(i);
}

// ---------------------------------------------------------------------------
// WMMA fragment load, 16-bit A-layout (ISA 7.12.2):
//   lane l -> matrix row (l & 15); lane[4] selects K-half.
//   lane holds two contiguous 8-element (16B) K-chunks:
//     [kb + hs*8, +8) and [kb + 16 + hs*8, +8)
// For C = A * W^T with W row-major (O x K), lane l of the B fragment reads
// row (col0 + (l&15)) of W -- identical addressing to A.
// ---------------------------------------------------------------------------
union ABFrag { u32x4 u[2]; v16bf v; };

__device__ __forceinline__ v16bf load_frag(const bf16* __restrict__ base,
                                           int ld, int row0, int kb, int lane) {
  int r  = lane & 15;
  int hs = lane >> 4;
  const bf16* p = base + (size_t)(row0 + r) * ld + kb + hs * 8;
  ABFrag f;
  f.u[0] = *(const u32x4*)(p);
  f.u[1] = *(const u32x4*)(p + 16);
  return f.v;
}

// ---------------------------------------------------------------------------
// C = A (RxK, bf16 row-major) * W^T (W is OxK, bf16 row-major), C f32 RxO.
// blockDim = 128 (4 waves); each wave computes two 16x16 output tiles.
// grid = (R/16, O/128).
// The 16x32 A tile (1KB) is shared by all 4 waves -> staged cooperatively
// into double-buffered LDS with gfx1250 async loads; B tiles are wave-
// private and stay on direct global_load_b128.
// ---------------------------------------------------------------------------
__global__ void gemm_bf16_wmma(const bf16* __restrict__ A,
                               const bf16* __restrict__ W,
                               float* __restrict__ C,
                               int R, int K, int O) {
  __shared__ bf16 shA[2][16 * 32];   // 2 x 1KB double buffer

  const int tid  = threadIdx.x;      // 128 threads = 4 waves
  const int lane = tid & 31;
  const int wave = tid >> 5;
  const int row0 = blockIdx.x * 16;
  const int col0 = (blockIdx.y * 8 + wave * 2) * 16;

  // Cooperative staging: each of 128 threads moves one 8B chunk of the 1KB
  // A tile (16 rows x 32 bf16).  arow = row in tile, achk = 8B chunk in row.
  const int arow = tid >> 3;         // 0..15
  const int achk = tid & 7;          // 0..7
  const size_t abase = (size_t)(row0 + arow) * K + achk * 4;

  auto issueA = [&](int buf, int kb) {
    const bf16* g = A + abase + kb;
    // Generic LDS pointers carry the byte offset in the low 32 bits.
    unsigned lds = (unsigned)(size_t)&shA[buf][arow * 32 + achk * 4];
    asm volatile("global_load_async_to_lds_b64 %0, %1, off"
                 :: "v"(lds), "v"(g) : "memory");
  };

  issueA(0, 0);

  v8f acc0 = {}; v8f acc1 = {};
  const int r_ = lane & 15, hs = lane >> 4;

  for (int kb = 0; kb < K; kb += 32) {
    const int cur = (kb >> 5) & 1;

    // Finish this wave's async writes, then make the tile visible block-wide.
    asm volatile("s_wait_asynccnt 0x0" ::: "memory");
    __syncthreads();

    // A fragment from LDS (two ds_load_b128 per lane).
    ABFrag fa;
    const bf16* ap = &shA[cur][r_ * 32 + hs * 8];
    fa.u[0] = *(const u32x4*)(ap);
    fa.u[1] = *(const u32x4*)(ap + 16);

    // Overlap: kick off the next A tile while this one is consumed.
    if (kb + 32 < K) issueA(cur ^ 1, kb + 32);
    if (kb + 64 < K)
      __builtin_prefetch((const void*)(W + (size_t)(col0 + r_) * K + kb + 64), 0, 1);

    v16bf b0 = load_frag(W, K, col0,      kb, lane);
    v16bf b1 = load_frag(W, K, col0 + 16, kb, lane);
    acc0 = __builtin_amdgcn_wmma_f32_16x16x32_bf16(false, fa.v, false, b0, (short)0, acc0, false, false);
    acc1 = __builtin_amdgcn_wmma_f32_16x16x32_bf16(false, fa.v, false, b1, (short)0, acc1, false, false);
  }

  // C/D layout: VGPR r, lanes 0-15 -> M=r, lanes 16-31 -> M=r+8; N = lane&15.
  const int n  = lane & 15;
  const int mb = (lane >> 4) * 8;
#pragma unroll
  for (int r = 0; r < 8; ++r) {
    const int m = mb + r;
    C[(size_t)(row0 + m) * O + col0 + n]      = acc0[r];
    C[(size_t)(row0 + m) * O + col0 + 16 + n] = acc1[r];
  }
}

// ---------------------------------------------------------------------------
// Elementwise helpers
// ---------------------------------------------------------------------------
__global__ void conv_f32_bf16_kernel(const float* __restrict__ in,
                                     bf16* __restrict__ out, size_t n) {
  size_t i = (size_t)blockIdx.x * blockDim.x + threadIdx.x;
  size_t st = (size_t)gridDim.x * blockDim.x;
  for (; i < n; i += st) out[i] = (bf16)in[i];
}

__global__ void fill_u32_kernel(unsigned* __restrict__ p, unsigned v, size_t n) {
  size_t i = (size_t)blockIdx.x * blockDim.x + threadIdx.x;
  size_t st = (size_t)gridDim.x * blockDim.x;
  for (; i < n; i += st) p[i] = v;
}

__global__ void add_f32_kernel(const float* __restrict__ a,
                               const float* __restrict__ b,
                               float* __restrict__ out, size_t n) {
  size_t i = (size_t)blockIdx.x * blockDim.x + threadIdx.x;
  size_t st = (size_t)gridDim.x * blockDim.x;
  for (; i < n; i += st) out[i] = a[i] + b[i];
}

__global__ void silu_mul_bf16_kernel(const float* __restrict__ g,
                                     const float* __restrict__ u,
                                     bf16* __restrict__ out, size_t n) {
  size_t i = (size_t)blockIdx.x * blockDim.x + threadIdx.x;
  size_t st = (size_t)gridDim.x * blockDim.x;
  for (; i < n; i += st) {
    float gv = g[i];
    out[i] = (bf16)((gv / (1.0f + __expf(-gv))) * u[i]);
  }
}

// ---------------------------------------------------------------------------
// Block-wide f32 sum reduction (wave32 shuffles + LDS across 8 waves)
// ---------------------------------------------------------------------------
__device__ __forceinline__ float block_reduce_sum(float v) {
  __shared__ float sh[32];
#pragma unroll
  for (int o = 16; o > 0; o >>= 1) v += __shfl_down(v, o, 32);
  const int lane = threadIdx.x & 31, w = threadIdx.x >> 5;
  if (lane == 0) sh[w] = v;
  __syncthreads();
  const int nw = blockDim.x >> 5;
  v = (threadIdx.x < (unsigned)nw) ? sh[threadIdx.x] : 0.0f;
  if (w == 0) {
#pragma unroll
    for (int o = 16; o > 0; o >>= 1) v += __shfl_down(v, o, 32);
    if (lane == 0) sh[0] = v;
  }
  __syncthreads();
  return sh[0];
}

// rmsnorm(x) * w -> bf16 out.  One block (256 thr) per row of HID_=1024.
__global__ void rmsnorm_bf16_kernel(const float* __restrict__ x,
                                    const float* __restrict__ w,
                                    bf16* __restrict__ out) {
  const size_t row = blockIdx.x;
  const float* xr = x + row * HID_;
  float v[4]; float ss = 0.0f;
#pragma unroll
  for (int i = 0; i < 4; ++i) { v[i] = xr[threadIdx.x + 256 * i]; ss += v[i] * v[i]; }
  ss = block_reduce_sum(ss);
  const float sc = rsqrtf(ss * (1.0f / HID_) + 1e-6f);
#pragma unroll
  for (int i = 0; i < 4; ++i) {
    const int c = threadIdx.x + 256 * i;
    out[row * HID_ + c] = (bf16)(w[c] * v[i] * sc);
  }
}

// h = a + b (stored f32); h2 = rmsnorm(h)*w -> bf16.
__global__ void add_rmsnorm_bf16_kernel(const float* __restrict__ a,
                                        const float* __restrict__ b,
                                        const float* __restrict__ w,
                                        float* __restrict__ h,
                                        bf16* __restrict__ out) {
  const size_t row = blockIdx.x;
  float v[4]; float ss = 0.0f;
#pragma unroll
  for (int i = 0; i < 4; ++i) {
    const size_t idx = row * HID_ + threadIdx.x + 256 * i;
    v[i] = a[idx] + b[idx];
    h[idx] = v[i];
    ss += v[i] * v[i];
  }
  ss = block_reduce_sum(ss);
  const float sc = rsqrtf(ss * (1.0f / HID_) + 1e-6f);
#pragma unroll
  for (int i = 0; i < 4; ++i) {
    const int c = threadIdx.x + 256 * i;
    out[row * HID_ + c] = (bf16)(w[c] * v[i] * sc);
  }
}

// ---------------------------------------------------------------------------
// RoPE tables: cos/sin for positions 0..2QL-1, dims 0..HD-1.
// ---------------------------------------------------------------------------
__global__ void rope_table_kernel(float* __restrict__ rc, float* __restrict__ rs) {
  const int idx = blockIdx.x * blockDim.x + threadIdx.x;
  if (idx >= KP_ * HD_) return;
  const int p = idx / HD_, d = idx % HD_;
  const float inv = __powf(10000.0f, -(float)(2 * (d & 31)) / 64.0f);
  const float ang = (float)p * inv;
  rc[idx] = __cosf(ang);
  rs[idx] = __sinf(ang);
}

// ---------------------------------------------------------------------------
// Pass A: per-edge RoPE'd logits + segment max (atomicMax, order-encoded).
// logits layout [E][NH][QL][KP].  Key position 2t = k[src], 2t+1 = ek[e].
// grid.x = E, blockDim = 256 (each thread 8 of the 2048 dot products).
// ---------------------------------------------------------------------------
__global__ void attn_logits_kernel(const float* __restrict__ q,
                                   const float* __restrict__ kbuf,
                                   const float* __restrict__ ekbuf,
                                   const int* __restrict__ src,
                                   const int* __restrict__ dst,
                                   const float* __restrict__ rc,
                                   const float* __restrict__ rs,
                                   float* __restrict__ logits,
                                   unsigned* __restrict__ maxbuf) {
  const int e = blockIdx.x;
  const int s = src[e], d = dst[e];
  for (int item = threadIdx.x; item < NH_ * QL_ * KP_; item += blockDim.x) {
    const int kpos = item & 15;
    const int qi   = (item >> 4) & 7;
    const int h    = item >> 7;
    const int kv   = h >> 1;                       // GQA: groups = NH/NKV = 2
    const int t    = kpos >> 1;
    const float* qr = q + ((size_t)d * QL_ + qi) * (NH_ * HD_) + h * HD_;
    const float* kr = (kpos & 1)
        ? ekbuf + ((size_t)e * QL_ + t) * (NKV_ * HD_) + kv * HD_
        : kbuf  + ((size_t)s * QL_ + t) * (NKV_ * HD_) + kv * HD_;
    const float* cq = rc + qi * HD_;  const float* sq = rs + qi * HD_;
    const float* ck = rc + kpos * HD_; const float* sk = rs + kpos * HD_;
    float acc = 0.0f;
#pragma unroll 8
    for (int dd = 0; dd < HD_; ++dd) {
      const float qv = qr[dd];
      const float qrot = (dd < 32) ? -qr[dd + 32] : qr[dd - 32];
      const float kvv = kr[dd];
      const float krot = (dd < 32) ? -kr[dd + 32] : kr[dd - 32];
      acc += (qv * cq[dd] + qrot * sq[dd]) * (kvv * ck[dd] + krot * sk[dd]);
    }
    acc *= 0.125f;                                  // 1/sqrt(HD)
    logits[(((size_t)e * NH_ + h) * QL_ + qi) * KP_ + kpos] = acc;
    atomicMax(&maxbuf[((size_t)d * NH_ + h) * QL_ + qi], enc_ord(acc));
  }
}

// Pass B: ex = exp(l - m), stored in place; segment sum via atomicAdd.
__global__ void attn_expsum_kernel(float* __restrict__ logits,
                                   const int* __restrict__ dst,
                                   const unsigned* __restrict__ maxbuf,
                                   float* __restrict__ sumbuf) {
  const size_t total = (size_t)E_ * NH_ * QL_ * KP_;
  size_t i = (size_t)blockIdx.x * blockDim.x + threadIdx.x;
  const size_t st = (size_t)gridDim.x * blockDim.x;
  for (; i < total; i += st) {
    size_t r = i >> 4;
    const int qi = (int)(r & 7); r >>= 3;
    const int h  = (int)(r & 15);
    const int e  = (int)(r >> 4);
    const int n  = dst[e];
    const float m = dec_ord(maxbuf[((size_t)n * NH_ + h) * QL_ + qi]);
    const float ex = __expf(logits[i] - m);
    logits[i] = ex;
    atomicAdd(&sumbuf[((size_t)n * NH_ + h) * QL_ + qi], ex);
  }
}

// Pass C: out_e = sum_k alpha * v; scatter-add into agg[dst].
// grid = (E, NH), blockDim = 64 (thread = head dim).  LDS stages ex + sums.
__global__ void attn_out_kernel(const float* __restrict__ logits,  // holds ex
                                const float* __restrict__ vbuf,
                                const float* __restrict__ evbuf,
                                const int* __restrict__ src,
                                const int* __restrict__ dst,
                                const float* __restrict__ sumbuf,
                                float* __restrict__ agg) {
  const int e = blockIdx.x, h = blockIdx.y;
  const int s = src[e], n = dst[e], kv = h >> 1;
  __shared__ float shex[QL_ * KP_];
  __shared__ float shs[QL_];
  const float* exb = logits + ((size_t)e * NH_ + h) * QL_ * KP_;
  for (int i = threadIdx.x; i < QL_ * KP_; i += blockDim.x) shex[i] = exb[i];
  if (threadIdx.x < QL_)
    shs[threadIdx.x] = sumbuf[((size_t)n * NH_ + h) * QL_ + threadIdx.x] + 1e-16f;
  __syncthreads();

  const int dd = threadIdx.x;   // 0..63
  float vv[KP_];
#pragma unroll
  for (int kpos = 0; kpos < KP_; ++kpos) {
    const int t = kpos >> 1;
    vv[kpos] = (kpos & 1)
        ? evbuf[((size_t)e * QL_ + t) * (NKV_ * HD_) + kv * HD_ + dd]
        : vbuf [((size_t)s * QL_ + t) * (NKV_ * HD_) + kv * HD_ + dd];
  }
#pragma unroll
  for (int qi = 0; qi < QL_; ++qi) {
    float acc = 0.0f;
#pragma unroll
    for (int kpos = 0; kpos < KP_; ++kpos) acc += shex[qi * KP_ + kpos] * vv[kpos];
    atomicAdd(&agg[((size_t)n * QL_ + qi) * HID_ + h * HD_ + dd], acc / shs[qi]);
  }
}

// ---------------------------------------------------------------------------
// Host-side orchestration
// ---------------------------------------------------------------------------
extern "C" void kernel_launch(void* const* d_in, const int* in_sizes, int n_in,
                              void* d_out, int out_size, void* d_ws, size_t ws_size,
                              hipStream_t stream) {
  (void)in_sizes; (void)n_in; (void)out_size; (void)ws_size;

  const float* hidden = (const float*)d_in[0];        // (N, QL, HID)
  const int*   eidx   = (const int*)  d_in[1];        // (2, E)
  const float* ehid   = (const float*)d_in[2];        // (E, QL, HID)
  const float* Wq     = (const float*)d_in[3];        // (NH*HD, HID)
  const float* Wk     = (const float*)d_in[4];        // (NKV*HD, HID)
  const float* Wv     = (const float*)d_in[5];
  const float* Wo     = (const float*)d_in[6];        // (HID, HID)
  const float* Wg     = (const float*)d_in[7];        // (INTER, HID)
  const float* Wu     = (const float*)d_in[8];
  const float* Wd     = (const float*)d_in[9];        // (HID, INTER)
  const float* ln1    = (const float*)d_in[10];
  const float* ln2    = (const float*)d_in[11];
  const int* src = eidx;
  const int* dst = eidx + E_;
  float* out = (float*)d_out;

  // ---- bump allocator over d_ws ----
  char* ws = (char*)d_ws;
  size_t off = 0;
  auto alloc = [&](size_t bytes) -> void* {
    off = (off + 255) & ~(size_t)255;
    void* p = ws + off;
    off += bytes;
    return p;
  };
  const size_t NR = (size_t)N_ * QL_;     // 8192 node rows
  const size_t ER = (size_t)E_ * QL_;     // 65536 edge rows

  bf16* xb   = (bf16*)alloc(NR * HID_ * 2);
  bf16* eb   = (bf16*)alloc(ER * HID_ * 2);
  bf16* wqb  = (bf16*)alloc((size_t)HID_ * HID_ * 2);
  bf16* wkb  = (bf16*)alloc((size_t)NKV_ * HD_ * HID_ * 2);
  bf16* wvb  = (bf16*)alloc((size_t)NKV_ * HD_ * HID_ * 2);
  bf16* wob  = (bf16*)alloc((size_t)HID_ * HID_ * 2);
  bf16* wgb  = (bf16*)alloc((size_t)INTER_ * HID_ * 2);
  bf16* wub  = (bf16*)alloc((size_t)INTER_ * HID_ * 2);
  bf16* wdb  = (bf16*)alloc((size_t)HID_ * INTER_ * 2);
  float* qbuf  = (float*)alloc(NR * HID_ * 4);
  float* kbuf  = (float*)alloc(NR * (NKV_ * HD_) * 4);
  float* vbuf  = (float*)alloc(NR * (NKV_ * HD_) * 4);
  float* ekbuf = (float*)alloc(ER * (NKV_ * HD_) * 4);
  float* evbuf = (float*)alloc(ER * (NKV_ * HD_) * 4);
  float* lg    = (float*)alloc((size_t)E_ * NH_ * QL_ * KP_ * 4);
  unsigned* maxb = (unsigned*)alloc((size_t)N_ * NH_ * QL_ * 4);
  float* sumb  = (float*)alloc((size_t)N_ * NH_ * QL_ * 4);
  float* agg   = (float*)alloc(NR * HID_ * 4);
  bf16*  aggb  = (bf16*)alloc(NR * HID_ * 2);
  float* attn  = (float*)alloc(NR * HID_ * 4);
  float* hbuf  = (float*)alloc(NR * HID_ * 4);
  bf16*  h2b   = (bf16*)alloc(NR * HID_ * 2);
  float* gate  = (float*)alloc(NR * INTER_ * 4);
  float* up    = (float*)alloc(NR * INTER_ * 4);
  bf16*  mb    = (bf16*)alloc(NR * INTER_ * 2);
  float* mlp   = (float*)alloc(NR * HID_ * 4);
  float* rc    = (float*)alloc((size_t)KP_ * HD_ * 4);
  float* rs    = (float*)alloc((size_t)KP_ * HD_ * 4);

  auto conv = [&](const float* in, bf16* o, size_t n) {
    conv_f32_bf16_kernel<<<4096, 256, 0, stream>>>(in, o, n);
  };
  auto gemm = [&](const bf16* A, const bf16* W, float* C, int R, int K, int O) {
    dim3 grid(R / 16, O / 128);
    gemm_bf16_wmma<<<grid, 128, 0, stream>>>(A, W, C, R, K, O);
  };

  // ---- 1. precision staging ----
  conv(Wq, wqb, (size_t)HID_ * HID_);
  conv(Wk, wkb, (size_t)NKV_ * HD_ * HID_);
  conv(Wv, wvb, (size_t)NKV_ * HD_ * HID_);
  conv(Wo, wob, (size_t)HID_ * HID_);
  conv(Wg, wgb, (size_t)INTER_ * HID_);
  conv(Wu, wub, (size_t)INTER_ * HID_);
  conv(Wd, wdb, (size_t)HID_ * INTER_);
  conv(ehid, eb, ER * HID_);                                    // edges: no norm
  rmsnorm_bf16_kernel<<<(int)NR, 256, 0, stream>>>(hidden, ln1, xb);
  rope_table_kernel<<<4, 256, 0, stream>>>(rc, rs);

  // ---- 2. projections (WMMA) ----
  gemm(xb, wqb, qbuf,  (int)NR, HID_, NH_ * HD_);               // q  8192x1024
  gemm(xb, wkb, kbuf,  (int)NR, HID_, NKV_ * HD_);              // k  8192x512
  gemm(xb, wvb, vbuf,  (int)NR, HID_, NKV_ * HD_);              // v
  gemm(eb, wkb, ekbuf, (int)ER, HID_, NKV_ * HD_);              // ek 65536x512
  gemm(eb, wvb, evbuf, (int)ER, HID_, NKV_ * HD_);              // ev

  // ---- 3. segment softmax attention ----
  fill_u32_kernel<<<512, 256, 0, stream>>>(maxb, 0u, (size_t)N_ * NH_ * QL_);
  fill_u32_kernel<<<512, 256, 0, stream>>>((unsigned*)sumb, 0u, (size_t)N_ * NH_ * QL_);
  fill_u32_kernel<<<4096, 256, 0, stream>>>((unsigned*)agg, 0u, NR * HID_);

  attn_logits_kernel<<<E_, 256, 0, stream>>>(qbuf, kbuf, ekbuf, src, dst, rc, rs, lg, maxb);
  attn_expsum_kernel<<<8192, 256, 0, stream>>>(lg, dst, maxb, sumb);
  attn_out_kernel<<<dim3(E_, NH_), 64, 0, stream>>>(lg, vbuf, evbuf, src, dst, sumb, agg);

  // ---- 4. output proj + residual + norm2 ----
  conv(agg, aggb, NR * HID_);
  gemm(aggb, wob, attn, (int)NR, HID_, HID_);
  add_rmsnorm_bf16_kernel<<<(int)NR, 256, 0, stream>>>(hidden, attn, ln2, hbuf, h2b);

  // ---- 5. MLP ----
  gemm(h2b, wgb, gate, (int)NR, HID_, INTER_);
  gemm(h2b, wub, up,   (int)NR, HID_, INTER_);
  silu_mul_bf16_kernel<<<8192, 256, 0, stream>>>(gate, up, mb, NR * INTER_);
  gemm(mb, wdb, mlp, (int)NR, INTER_, HID_);
  add_f32_kernel<<<4096, 256, 0, stream>>>(hbuf, mlp, out, NR * HID_);
}
